// OthelloGPT_47081431498814
// MI455X (gfx1250) — matrix operand order repaired and checked by reading, not compiled, not run on hardware
//
#include <hip/hip_runtime.h>
#include <hip/hip_bf16.h>
#include <math.h>

// ---------------- model dims ----------------
#define LNUM 8
#define DMODEL 512
#define HNUM 8
#define SLEN 64
#define VOCAB 66
#define BATCH 256
#define DHEAD 64
#define DFF 2048
#define NTOK (BATCH * SLEN)   // 16384

typedef __attribute__((ext_vector_type(16))) _Float16 v16h;
typedef __attribute__((ext_vector_type(8)))  _Float16 h8v;
typedef __attribute__((ext_vector_type(8)))  float    v8f;

union FragH { v16h v; h8v q2[2]; _Float16 h[16]; };

enum { MODE_F16 = 0, MODE_GELU_F16 = 1, MODE_RESID_F32 = 2, MODE_F32_BIAS = 3 };

static __device__ inline v8f wmma16(v16h a, v16h b, v8f c) {
  // emits v_wmma_f32_16x16x32_f16
  return __builtin_amdgcn_wmma_f32_16x16x32_f16(false, a, false, b, (short)0, c,
                                                false, false);
}

// A-fragment (16x32 f16): lane<16 holds K[k0..k0+15] of row
// (lane&15); lane>=16 shifted by +16 in K. p points at
// A + row*lda + k0 + ((lane>>4)*16). Two contiguous 16B loads.
static __device__ inline v16h load_a_frag(const _Float16* p) {
  FragH f;
  f.q2[0] = *(const h8v*)p;
  f.q2[1] = *(const h8v*)(p + 8);
  return f.v;
}

static __device__ inline float wred_sum(float v) {
  #pragma unroll
  for (int o = 16; o > 0; o >>= 1) v += __shfl_xor(v, o, 32);
  return v;
}
static __device__ inline float wred_max(float v) {
  #pragma unroll
  for (int o = 16; o > 0; o >>= 1) v = fmaxf(v, __shfl_xor(v, o, 32));
  return v;
}

// ------- weight swizzle: f32 [K,N] row-major -> f16 WMMA B-fragment order ---
// dst[((kb*NT + nb)*32 + lane)*16 + i] = B[(kb*32 + i + 16*(lane>=16))*N
//                                          + nb*16 + (lane&15)]
// (zero-padded past N). GEMM then loads each B fragment as one contiguous
// 32-byte chunk per lane. NT templated so div/mod strength-reduce.
template <int NT>
__global__ __launch_bounds__(256) void wswz_kernel(
    const float* __restrict__ src, _Float16* __restrict__ dst, int K, int N) {
  const int idx = blockIdx.x * 256 + threadIdx.x;
  const int i = idx & 15;
  const int lane = (idx >> 4) & 31;
  const int tile = idx >> 9;
  const int nb = tile % NT;
  const int kb = tile / NT;
  const int k = kb * 32 + i + ((lane >> 4) << 4);
  const int n = nb * 16 + (lane & 15);
  dst[idx] = (n < N) ? (_Float16)src[(size_t)k * N + n] : (_Float16)0.f;
}

// ---------------- WMMA GEMM: C[M,N] = A[M,K] x Bsw[K,Npad] (+bias) ----------
// A row-major f16; B pre-swizzled to fragment order (see wswz_kernel).
// K and NT (= Npad/16) are compile-time so all B addressing folds into
// 24-bit instruction offsets; MODE is compile-time so each instantiation
// carries exactly one epilogue path.
// 8 waves/block: wave (wm=wave>>2, wn=wave&3) computes a 32x32 block;
// block covers 64 rows x 128 cols. Inner loop: 4 b128-pair loads + 4 WMMA.
template <int MODE>
static __device__ inline void gemm_epilogue(v8f acc, int mrow, int n, int N,
                                            const float* bias, void* out) {
  if (n >= N) return;
  const float bv = (MODE == MODE_F16) ? 0.f : bias[n];
  #pragma unroll
  for (int r = 0; r < 8; ++r) {
    const size_t idx = (size_t)(mrow + r) * N + n;
    const float val = acc[r] + bv;
    if constexpr (MODE == MODE_F16) {
      ((_Float16*)out)[idx] = (_Float16)val;
    } else if constexpr (MODE == MODE_GELU_F16) {
      const float g = 0.5f * val * (1.f + erff(val * 0.70710678118654752440f));
      ((_Float16*)out)[idx] = (_Float16)g;
    } else if constexpr (MODE == MODE_RESID_F32) {
      ((float*)out)[idx] += val;
    } else {  // MODE_F32_BIAS
      ((float*)out)[idx] = val;
    }
  }
}

template <int K, int NT, int MODE>
__global__ __launch_bounds__(256) void gemm_wmma_kernel(
    const _Float16* __restrict__ A, const _Float16* __restrict__ B,
    const float* __restrict__ bias, void* out, int M, int N) {
  const int lane = threadIdx.x & 31;
  const int wave = threadIdx.x >> 5;
  const int wm = wave >> 2;                 // 0..1
  const int wn = wave & 3;                  // 0..3
  const int m0 = blockIdx.y * 64 + wm * 32;
  const int nb0raw = blockIdx.x * 8 + wn * 2;
  const int nb0 = min(nb0raw, NT - 1);      // clamp: loads stay in-bounds,
  const int nb1 = min(nb0raw + 1, NT - 1);  // stores are guarded below

  v8f acc00 = {}, acc01 = {}, acc10 = {}, acc11 = {};
  const _Float16* arow0 =
      A + (size_t)(m0 + (lane & 15)) * K + ((lane >> 4) << 4);
  const _Float16* arow1 = arow0 + (size_t)16 * K;
  const _Float16* b0p = B + ((size_t)nb0 * 32 + lane) * 16;
  const _Float16* b1p = B + ((size_t)nb1 * 32 + lane) * 16;

  #pragma unroll 4
  for (int kb = 0; kb < K / 32; ++kb) {
    v16h a0 = load_a_frag(arow0 + kb * 32);
    v16h a1 = load_a_frag(arow1 + kb * 32);
    FragH b0, b1;
    const int boff = kb * NT * 512;  // constant-coefficient -> imm offsets
    b0.q2[0] = *(const h8v*)(b0p + boff);
    b0.q2[1] = *(const h8v*)(b0p + boff + 8);
    b1.q2[0] = *(const h8v*)(b1p + boff);
    b1.q2[1] = *(const h8v*)(b1p + boff + 8);
    acc00 = wmma16(a0, b0.v, acc00);
    acc01 = wmma16(a0, b1.v, acc01);
    acc10 = wmma16(a1, b0.v, acc10);
    acc11 = wmma16(a1, b1.v, acc11);
  }
  const int mrow0 = m0 + ((lane >> 4) << 3);
  const int mrow1 = mrow0 + 16;
  const int n0 = nb0raw * 16 + (lane & 15);
  const int n1 = n0 + 16;
  gemm_epilogue<MODE>(acc00, mrow0, n0, N, bias, out);
  gemm_epilogue<MODE>(acc01, mrow0, n1, N, bias, out);
  gemm_epilogue<MODE>(acc10, mrow1, n0, N, bias, out);
  gemm_epilogue<MODE>(acc11, mrow1, n1, N, bias, out);
}

// ---------------- embedding ----------------
__global__ __launch_bounds__(256) void embed_kernel(
    const int* __restrict__ ids, const float* __restrict__ tok,
    const float* __restrict__ pos, float* __restrict__ x) {
  const int row = blockIdx.x;          // token index b*S+s
  const int s = row & (SLEN - 1);
  const int id = ids[row];
  for (int d = threadIdx.x; d < DMODEL; d += 256)
    x[(size_t)row * DMODEL + d] =
        tok[(size_t)id * DMODEL + d] + pos[(size_t)s * DMODEL + d];
}

// ---------------- layernorm (unbiased std, eps on std) ----------------
__global__ __launch_bounds__(256) void ln_kernel(
    const float* __restrict__ x, const float* __restrict__ g,
    const float* __restrict__ b, _Float16* __restrict__ out) {
  const int lane = threadIdx.x & 31;
  const int wave = threadIdx.x >> 5;
  const int row = blockIdx.x * 8 + wave;
  const float* xr = x + (size_t)row * DMODEL;
  float v[16];
  float s = 0.f;
  #pragma unroll
  for (int j = 0; j < 16; ++j) { v[j] = xr[lane + j * 32]; s += v[j]; }
  const float mean = wred_sum(s) * (1.f / DMODEL);
  float ss = 0.f;
  #pragma unroll
  for (int j = 0; j < 16; ++j) { float d = v[j] - mean; ss += d * d; }
  const float var = wred_sum(ss) * (1.f / (DMODEL - 1));
  const float inv = 1.f / (sqrtf(var) + 1e-10f);
  #pragma unroll
  for (int j = 0; j < 16; ++j) {
    const int e = lane + j * 32;
    out[(size_t)row * DMODEL + e] =
        (_Float16)((v[j] - mean) * inv * g[e] + b[e]);
  }
}

// ---------------- fused causal attention per (b,h) ----------------
__global__ __launch_bounds__(256) void attn_kernel(
    const _Float16* __restrict__ qg, const _Float16* __restrict__ kg,
    const _Float16* __restrict__ vg, _Float16* __restrict__ og) {
  __shared__ _Float16 sq[SLEN * DHEAD];      // q, later attn probs
  __shared__ _Float16 sk[SLEN * DHEAD];      // k (row-major; k^T frags are
                                             // contiguous reads)
  __shared__ _Float16 svt[DHEAD * SLEN];     // v transposed: svt[e*64 + t]
  __shared__ float ss[SLEN * SLEN];

  const int b = blockIdx.x >> 3;
  const int h = blockIdx.x & 7;
  const int tid = threadIdx.x;
  const int lane = tid & 31;
  const int wave = tid >> 5;
  const float scale = 0.044194173824159216f;  // 1/sqrt(512)

  // stage q,k via async global->LDS DMA (ASYNCcnt); v transposed via VGPRs
  {
    const int row = tid >> 2;
    const int quad = (tid & 3) * 16;
    const size_t src = ((size_t)(b * SLEN + row)) * DMODEL + h * DHEAD + quad;
    const int dst = row * DHEAD + quad;
    const unsigned lq = (unsigned)(uintptr_t)&sq[dst];
    const unsigned lk = (unsigned)(uintptr_t)&sk[dst];
    const _Float16* gq = &qg[src];
    const _Float16* gk = &kg[src];
    // inst offset applies to both sides.
    asm volatile(
        "global_load_async_to_lds_b128 %0, %2, off\n\t"
        "global_load_async_to_lds_b128 %0, %2, off offset:16\n\t"
        "global_load_async_to_lds_b128 %1, %3, off\n\t"
        "global_load_async_to_lds_b128 %1, %3, off offset:16"
        :
        : "v"(lq), "v"(lk), "v"(gq), "v"(gk)
        : "memory");
    FragH tv;
    tv.q2[0] = *(const h8v*)&vg[src];
    tv.q2[1] = *(const h8v*)&vg[src + 8];
    #pragma unroll
    for (int j = 0; j < 16; ++j) svt[(quad + j) * SLEN + row] = tv.h[j];
    asm volatile("s_wait_asynccnt 0x0" ::: "memory");
  }
  __syncthreads();

  const int m0 = (wave >> 1) * 16;
  const _Float16* arow =
      sq + (m0 + (lane & 15)) * DHEAD + ((lane >> 4) << 4);

  // scores = q @ k^T
  #pragma unroll
  for (int t = 0; t < 2; ++t) {
    const int n0 = ((wave & 1) * 2 + t) * 16;
    v8f acc = {};
    #pragma unroll
    for (int k0 = 0; k0 < DHEAD; k0 += 32) {
      v16h a = load_a_frag(arow + k0);
      const _Float16* p =
          sk + (n0 + (lane & 15)) * DHEAD + k0 + ((lane >> 4) << 4);
      FragH f;
      f.q2[0] = *(const h8v*)p;
      f.q2[1] = *(const h8v*)(p + 8);
      acc = wmma16(a, f.v, acc);
    }
    const int n = n0 + (lane & 15);
    const int mbase = m0 + ((lane >> 4) << 3);
    #pragma unroll
    for (int r = 0; r < 8; ++r) {
      const int m = mbase + r;
      ss[m * SLEN + n] = (n <= m) ? acc[r] * scale : -__builtin_inff();
    }
  }
  __syncthreads();

  // softmax rows -> attn probs (f16) overwriting sq
  #pragma unroll
  for (int rr = 0; rr < 8; ++rr) {
    const int row = wave * 8 + rr;
    const float v0 = ss[row * SLEN + lane];
    const float v1 = ss[row * SLEN + lane + 32];
    const float mx = wred_max(fmaxf(v0, v1));
    const float e0 = expf(v0 - mx);
    const float e1 = expf(v1 - mx);
    const float inv = 1.f / wred_sum(e0 + e1);
    sq[row * SLEN + lane] = (_Float16)(e0 * inv);
    sq[row * SLEN + lane + 32] = (_Float16)(e1 * inv);
  }
  __syncthreads();

  // o = attn @ v ; v staged transposed so B frags are contiguous LDS reads
  #pragma unroll
  for (int t = 0; t < 2; ++t) {
    const int n0 = ((wave & 1) * 2 + t) * 16;
    const int n = n0 + (lane & 15);
    v8f acc = {};
    #pragma unroll
    for (int k0 = 0; k0 < SLEN; k0 += 32) {
      v16h a = load_a_frag(arow + k0);
      const _Float16* p = svt + n * SLEN + k0 + ((lane >> 4) << 4);
      FragH f;
      f.q2[0] = *(const h8v*)p;
      f.q2[1] = *(const h8v*)(p + 8);
      acc = wmma16(a, f.v, acc);
    }
    const int mbase = m0 + ((lane >> 4) << 3);
    #pragma unroll
    for (int r = 0; r < 8; ++r) {
      const int m = mbase + r;
      og[((size_t)(b * SLEN + m)) * DMODEL + h * DHEAD + n] = (_Float16)acc[r];
    }
  }
}

// ---------------- loss: per-row log-softmax NLL, deterministic reduce ------
__global__ __launch_bounds__(256) void loss_row_kernel(
    const float* __restrict__ logits, const int* __restrict__ targets,
    float* __restrict__ buf) {
  const int lane = threadIdx.x & 31;
  const int wave = threadIdx.x >> 5;
  const int row = blockIdx.x * 8 + wave;
  const float* lr = logits + (size_t)row * VOCAB;
  const float v0 = lr[lane];
  const float v1 = lr[lane + 32];
  const float v2 = (lane + 64 < VOCAB) ? lr[lane + 64] : -__builtin_inff();
  const float mx = wred_max(fmaxf(fmaxf(v0, v1), v2));
  float s = expf(v0 - mx) + expf(v1 - mx) + expf(v2 - mx);
  s = wred_sum(s);
  if (lane == 0) buf[row] = logf(s) + mx - lr[targets[row]];
}

__global__ __launch_bounds__(256) void loss_reduce_kernel(
    const float* __restrict__ buf, float* __restrict__ out) {
  __shared__ float sm[256];
  float s = 0.f;
  for (int i = threadIdx.x; i < NTOK; i += 256) s += buf[i];
  sm[threadIdx.x] = s;
  __syncthreads();
  for (int st = 128; st > 0; st >>= 1) {
    if ((int)threadIdx.x < st) sm[threadIdx.x] += sm[threadIdx.x + st];
    __syncthreads();
  }
  if (threadIdx.x == 0) *out = sm[0] * (1.f / NTOK);
}

// ---------------- host orchestration ----------------
extern "C" void kernel_launch(void* const* d_in, const int* in_sizes, int n_in,
                              void* d_out, int out_size, void* d_ws,
                              size_t ws_size, hipStream_t stream) {
  const int*   ids    = (const int*)d_in[0];
  const int*   tgts   = (const int*)d_in[1];
  const float* tok    = (const float*)d_in[2];
  const float* pos    = (const float*)d_in[3];
  const float* Wq     = (const float*)d_in[4];
  const float* Wk     = (const float*)d_in[5];
  const float* Wv     = (const float*)d_in[6];
  const float* Wo     = (const float*)d_in[7];
  const float* bo     = (const float*)d_in[8];
  const float* W1     = (const float*)d_in[9];
  const float* b1     = (const float*)d_in[10];
  const float* W2     = (const float*)d_in[11];
  const float* b2     = (const float*)d_in[12];
  const float* ln1g   = (const float*)d_in[13];
  const float* ln1b   = (const float*)d_in[14];
  const float* ln2g   = (const float*)d_in[15];
  const float* ln2b   = (const float*)d_in[16];
  const float* lnfg   = (const float*)d_in[17];
  const float* lnfb   = (const float*)d_in[18];
  const float* Wu     = (const float*)d_in[19];
  const float* bu     = (const float*)d_in[20];

  float* logits = (float*)d_out;                 // NTOK x VOCAB
  float* loss   = logits + (size_t)NTOK * VOCAB; // scalar

  // workspace carve-up (256B aligned chunks)
  size_t off = 0;
  auto carve = [&](size_t bytes) -> void* {
    void* p = (char*)d_ws + off;
    off += (bytes + 255) & ~(size_t)255;
    return p;
  };
  // swizzled-weight sizes (halves): per layer
  const size_t szDD = (size_t)(DMODEL / 32) * (DMODEL / 16) * 512;  // 262144
  const size_t szDF = (size_t)(DMODEL / 32) * (DFF / 16) * 512;     // 1048576
  const size_t szFD = (size_t)(DFF / 32) * (DMODEL / 16) * 512;     // 1048576
  const int    nTu  = (VOCAB + 15) / 16;                            // 5
  const size_t szDU = (size_t)(DMODEL / 32) * nTu * 512;            // 40960

  float*    xf    = (float*)carve((size_t)NTOK * DMODEL * 4);
  _Float16* hf    = (_Float16*)carve((size_t)NTOK * DMODEL * 2);
  _Float16* qf    = (_Float16*)carve((size_t)NTOK * DMODEL * 2);
  _Float16* kf    = (_Float16*)carve((size_t)NTOK * DMODEL * 2);
  _Float16* vf    = (_Float16*)carve((size_t)NTOK * DMODEL * 2);
  _Float16* of    = (_Float16*)carve((size_t)NTOK * DMODEL * 2);
  _Float16* hidf  = (_Float16*)carve((size_t)NTOK * DFF * 2);
  _Float16* wqs   = (_Float16*)carve(LNUM * szDD * 2);
  _Float16* wks   = (_Float16*)carve(LNUM * szDD * 2);
  _Float16* wvs   = (_Float16*)carve(LNUM * szDD * 2);
  _Float16* wos   = (_Float16*)carve(LNUM * szDD * 2);
  _Float16* w1s   = (_Float16*)carve(LNUM * szDF * 2);
  _Float16* w2s   = (_Float16*)carve(LNUM * szFD * 2);
  _Float16* wus   = (_Float16*)carve(szDU * 2);
  float*    lbuf  = (float*)carve((size_t)NTOK * 4);

  for (int l = 0; l < LNUM; ++l) {
    const size_t oDD = (size_t)l * DMODEL * DMODEL;
    // (K/32)*NT*512/256 blocks each
    wswz_kernel<32><<<1024, 256, 0, stream>>>(Wq + oDD, wqs + l * szDD,
                                              DMODEL, DMODEL);
    wswz_kernel<32><<<1024, 256, 0, stream>>>(Wk + oDD, wks + l * szDD,
                                              DMODEL, DMODEL);
    wswz_kernel<32><<<1024, 256, 0, stream>>>(Wv + oDD, wvs + l * szDD,
                                              DMODEL, DMODEL);
    wswz_kernel<32><<<1024, 256, 0, stream>>>(Wo + oDD, wos + l * szDD,
                                              DMODEL, DMODEL);
    wswz_kernel<128><<<4096, 256, 0, stream>>>(
        W1 + (size_t)l * DMODEL * DFF, w1s + l * szDF, DMODEL, DFF);
    wswz_kernel<32><<<4096, 256, 0, stream>>>(
        W2 + (size_t)l * DFF * DMODEL, w2s + l * szFD, DFF, DMODEL);
  }
  wswz_kernel<5><<<160, 256, 0, stream>>>(Wu, wus, DMODEL, VOCAB);

  embed_kernel<<<NTOK, 256, 0, stream>>>(ids, tok, pos, xf);

  const dim3 gDD(4, NTOK / 64);    // N=512  (NT=32)
  const dim3 gDF(16, NTOK / 64);   // N=2048 (NT=128)
  const dim3 gDU(1, NTOK / 64);    // N=66   (NT=5)

  for (int l = 0; l < LNUM; ++l) {
    ln_kernel<<<NTOK / 8, 256, 0, stream>>>(xf, ln1g + l * DMODEL,
                                            ln1b + l * DMODEL, hf);
    gemm_wmma_kernel<512, 32, MODE_F16><<<gDD, 256, 0, stream>>>(
        hf, wqs + l * szDD, nullptr, qf, NTOK, DMODEL);
    gemm_wmma_kernel<512, 32, MODE_F16><<<gDD, 256, 0, stream>>>(
        hf, wks + l * szDD, nullptr, kf, NTOK, DMODEL);
    gemm_wmma_kernel<512, 32, MODE_F16><<<gDD, 256, 0, stream>>>(
        hf, wvs + l * szDD, nullptr, vf, NTOK, DMODEL);
    attn_kernel<<<BATCH * HNUM, 256, 0, stream>>>(qf, kf, vf, of);
    gemm_wmma_kernel<512, 32, MODE_RESID_F32><<<gDD, 256, 0, stream>>>(
        of, wos + l * szDD, bo + l * DMODEL, xf, NTOK, DMODEL);
    ln_kernel<<<NTOK / 8, 256, 0, stream>>>(xf, ln2g + l * DMODEL,
                                            ln2b + l * DMODEL, hf);
    gemm_wmma_kernel<512, 128, MODE_GELU_F16><<<gDF, 256, 0, stream>>>(
        hf, w1s + l * szDF, b1 + l * DFF, hidf, NTOK, DFF);
    gemm_wmma_kernel<2048, 32, MODE_RESID_F32><<<gDD, 256, 0, stream>>>(
        hidf, w2s + l * szFD, b2 + l * DMODEL, xf, NTOK, DMODEL);
  }

  ln_kernel<<<NTOK / 8, 256, 0, stream>>>(xf, lnfg, lnfb, hf);
  gemm_wmma_kernel<512, 5, MODE_F32_BIAS><<<gDU, 256, 0, stream>>>(
      hf, wus, bu, logits, NTOK, VOCAB);
  loss_row_kernel<<<NTOK / 8, 256, 0, stream>>>(logits, tgts, lbuf);
  loss_reduce_kernel<<<1, 256, 0, stream>>>(lbuf, loss);
}
